// ResContextBlock_14070312862109
// MI455X (gfx1250) — compile-verified, matrix operand order checked by reading, not run on hardware
//
#include <hip/hip_runtime.h>

typedef __attribute__((ext_vector_type(16))) _Float16 v16h;
typedef __attribute__((ext_vector_type(8)))  float    v8f;

#define ZD 32
#define YD 360
#define XD 480
#define GRID_SZ (ZD * YD * XD)
#define NVOX 200000
#define NTILES (NVOX / 16)

// ---------------- hash table ----------------

__global__ void fill_table_k(int* __restrict__ t) {
  int i = blockIdx.x * blockDim.x + threadIdx.x;
  if (i < GRID_SZ) t[i] = -1;
}

__global__ void scatter_table_k(const int* __restrict__ coords, int* __restrict__ t) {
  int n = blockIdx.x * blockDim.x + threadIdx.x;
  if (n >= NVOX) return;
  int z = coords[3 * n + 0], y = coords[3 * n + 1], x = coords[3 * n + 2];
  t[(z * YD + y) * XD + x] = n;
}

// neighbor index maps for both kernel patterns, computed once and reused twice each
__global__ void nbidx_k(const int* __restrict__ coords, const int* __restrict__ t,
                        int* __restrict__ i133, int* __restrict__ i313) {
  int n = blockIdx.x * blockDim.x + threadIdx.x;
  if (n >= NVOX) return;
  int z = coords[3 * n + 0], y = coords[3 * n + 1], x = coords[3 * n + 2];
#pragma unroll
  for (int k = 0; k < 9; ++k) {
    int da = k / 3 - 1, dx = k % 3 - 1;
    int xx = x + dx;
    int yy = y + da;
    int id = -1;
    if (yy >= 0 && yy < YD && xx >= 0 && xx < XD)
      id = t[(z * YD + yy) * XD + xx];
    i133[k * NVOX + n] = id;
    int zz = z + da;
    id = -1;
    if (zz >= 0 && zz < ZD && xx >= 0 && xx < XD)
      id = t[(zz * YD + y) * XD + xx];
    i313[k * NVOX + n] = id;
  }
}

// ---------------- weight packing into WMMA B-fragment layout ----------------
// K index of the j-th half held by `lane` within a 32-deep K chunk
// (per CDNA5 ISA 16-bit A/B layouts: lanes<16 hold K{0-7,16-23}+0, lanes>=16 +8)
__device__ __forceinline__ int klocal(int lane, int j) {
  return ((j < 8) ? 0 : 16) + (j & 7) + ((lane >> 4) << 3);
}

// W: [9][16][32]  ->  out: [5 chunks][2 ntiles][32 lanes][16 halves], K padded 144->160
__global__ void pack_w16_k(const float* __restrict__ W, _Float16* __restrict__ out) {
  int tid = blockIdx.x * blockDim.x + threadIdx.x;
  if (tid >= 5 * 2 * 32) return;
  int lane = tid & 31, nt = (tid >> 5) & 1, chunk = tid >> 6;
  int col = (lane & 15) + nt * 16;
#pragma unroll
  for (int j = 0; j < 16; ++j) {
    int K = chunk * 32 + klocal(lane, j);
    int k = K >> 4, c = K & 15;
    float w = (k < 9) ? W[(k * 16 + c) * 32 + col] : 0.0f;
    out[((chunk * 2 + nt) * 32 + lane) * 16 + j] = (_Float16)w;
  }
}

// W: [9][32][32]  ->  out: [9 chunks][2 ntiles][32 lanes][16 halves]
__global__ void pack_w32_k(const float* __restrict__ W, _Float16* __restrict__ out) {
  int tid = blockIdx.x * blockDim.x + threadIdx.x;
  if (tid >= 9 * 2 * 32) return;
  int lane = tid & 31, nt = (tid >> 5) & 1, chunk = tid >> 6;
  int col = (lane & 15) + nt * 16;
#pragma unroll
  for (int j = 0; j < 16; ++j) {
    int c = klocal(lane, j);
    float w = W[(chunk * 32 + c) * 32 + col];
    out[((chunk * 2 + nt) * 32 + lane) * 16 + j] = (_Float16)w;
  }
}

// ---------------- fused epilogue: LeakyReLU -> BN -> (+addend) -> store ----------------
__device__ __forceinline__ void epilogue(v8f acc, int tile, int base, int lane,
                                         const float* __restrict__ gm,
                                         const float* __restrict__ bt,
                                         const float* __restrict__ mu,
                                         const float* __restrict__ vr, int layer,
                                         const float* __restrict__ addend,
                                         float* __restrict__ out) {
  int col = (lane & 15) + tile * 16;
  int hi = lane >> 4;
  float sc = gm[layer * 32 + col] * rsqrtf(vr[layer * 32 + col] + 1e-5f);
  float sh = bt[layer * 32 + col] - mu[layer * 32 + col] * sc;
#pragma unroll
  for (int v = 0; v < 8; ++v) {
    int m = base + v + hi * 8;
    float x = acc[v];
    x = (x >= 0.0f) ? x : 0.01f * x;
    x = x * sc + sh;
    int o = m * 32 + col;
    if (addend) x += addend[o];
    out[o] = x;
  }
}

// ---------------- conv, Cin=16 (K = 9*16 padded to 160 = 5 chunks of 32) ----------------
__global__ void __launch_bounds__(256)
conv16_k(const float* __restrict__ feats, const int* __restrict__ nb,
         const _Float16* __restrict__ wp,
         const float* __restrict__ gm, const float* __restrict__ bt,
         const float* __restrict__ mu, const float* __restrict__ vr, int layer,
         float* __restrict__ out) {
  int lane = threadIdx.x & 31;
  int wid = blockIdx.x * (blockDim.x >> 5) + (threadIdx.x >> 5);
  if (wid >= NTILES) return;
  int base = wid * 16;
  int row = lane & 15;
  int cb = (lane >> 4) * 8;  // channel base for this lane's K-halves

  v8f acc0 = {}; v8f acc1 = {};
#pragma unroll
  for (int chunk = 0; chunk < 5; ++chunk) {
    int k0 = chunk * 2, k1 = k0 + 1;
    int vi0 = nb[k0 * NVOX + base + row];
    int vi1 = (k1 < 9) ? nb[k1 * NVOX + base + row] : -1;
    v16h a = {};
    if (vi0 >= 0) {
      const float4* p = (const float4*)(feats + vi0 * 16 + cb);
      float4 u = p[0], w = p[1];
      a[0] = (_Float16)u.x; a[1] = (_Float16)u.y; a[2] = (_Float16)u.z; a[3] = (_Float16)u.w;
      a[4] = (_Float16)w.x; a[5] = (_Float16)w.y; a[6] = (_Float16)w.z; a[7] = (_Float16)w.w;
    }
    if (vi1 >= 0) {
      const float4* p = (const float4*)(feats + vi1 * 16 + cb);
      float4 u = p[0], w = p[1];
      a[8]  = (_Float16)u.x; a[9]  = (_Float16)u.y; a[10] = (_Float16)u.z; a[11] = (_Float16)u.w;
      a[12] = (_Float16)w.x; a[13] = (_Float16)w.y; a[14] = (_Float16)w.z; a[15] = (_Float16)w.w;
    }
    v16h b0 = *(const v16h*)(wp + ((chunk * 2 + 0) * 32 + lane) * 16);
    v16h b1 = *(const v16h*)(wp + ((chunk * 2 + 1) * 32 + lane) * 16);
    acc0 = __builtin_amdgcn_wmma_f32_16x16x32_f16(false, a, false, b0, (short)0, acc0, false, false);
    acc1 = __builtin_amdgcn_wmma_f32_16x16x32_f16(false, a, false, b1, (short)0, acc1, false, false);
  }
  epilogue(acc0, 0, base, lane, gm, bt, mu, vr, layer, nullptr, out);
  epilogue(acc1, 1, base, lane, gm, bt, mu, vr, layer, nullptr, out);
}

// ---------------- conv, Cin=32 (K = 9*32 = 9 chunks of 32), optional residual add ----------------
__global__ void __launch_bounds__(256)
conv32_k(const float* __restrict__ fin, const int* __restrict__ nb,
         const _Float16* __restrict__ wp,
         const float* __restrict__ gm, const float* __restrict__ bt,
         const float* __restrict__ mu, const float* __restrict__ vr, int layer,
         const float* __restrict__ addend, float* __restrict__ out) {
  int lane = threadIdx.x & 31;
  int wid = blockIdx.x * (blockDim.x >> 5) + (threadIdx.x >> 5);
  if (wid >= NTILES) return;
  int base = wid * 16;
  int row = lane & 15;
  int cb = (lane >> 4) * 8;

  v8f acc0 = {}; v8f acc1 = {};
#pragma unroll
  for (int k = 0; k < 9; ++k) {
    int vi = nb[k * NVOX + base + row];
    v16h a = {};
    if (vi >= 0) {
      const float* p = fin + vi * 32 + cb;
      float4 u0 = *(const float4*)(p);
      float4 u1 = *(const float4*)(p + 4);
      float4 u2 = *(const float4*)(p + 16);
      float4 u3 = *(const float4*)(p + 20);
      a[0]  = (_Float16)u0.x; a[1]  = (_Float16)u0.y; a[2]  = (_Float16)u0.z; a[3]  = (_Float16)u0.w;
      a[4]  = (_Float16)u1.x; a[5]  = (_Float16)u1.y; a[6]  = (_Float16)u1.z; a[7]  = (_Float16)u1.w;
      a[8]  = (_Float16)u2.x; a[9]  = (_Float16)u2.y; a[10] = (_Float16)u2.z; a[11] = (_Float16)u2.w;
      a[12] = (_Float16)u3.x; a[13] = (_Float16)u3.y; a[14] = (_Float16)u3.z; a[15] = (_Float16)u3.w;
    }
    v16h b0 = *(const v16h*)(wp + ((k * 2 + 0) * 32 + lane) * 16);
    v16h b1 = *(const v16h*)(wp + ((k * 2 + 1) * 32 + lane) * 16);
    acc0 = __builtin_amdgcn_wmma_f32_16x16x32_f16(false, a, false, b0, (short)0, acc0, false, false);
    acc1 = __builtin_amdgcn_wmma_f32_16x16x32_f16(false, a, false, b1, (short)0, acc1, false, false);
  }
  epilogue(acc0, 0, base, lane, gm, bt, mu, vr, layer, addend, out);
  epilogue(acc1, 1, base, lane, gm, bt, mu, vr, layer, addend, out);
}

// ---------------- host side ----------------

extern "C" void kernel_launch(void* const* d_in, const int* in_sizes, int n_in,
                              void* d_out, int out_size, void* d_ws, size_t ws_size,
                              hipStream_t stream) {
  const float* feats = (const float*)d_in[0];
  const int*   coords = (const int*)d_in[1];
  const float* w1  = (const float*)d_in[2];
  const float* w12 = (const float*)d_in[3];
  const float* w2  = (const float*)d_in[4];
  const float* w3  = (const float*)d_in[5];
  const float* gm  = (const float*)d_in[6];
  const float* bt  = (const float*)d_in[7];
  const float* mu  = (const float*)d_in[8];
  const float* vr  = (const float*)d_in[9];
  (void)in_sizes; (void)n_in; (void)out_size; (void)ws_size;

  char* ws = (char*)d_ws;
  size_t off = 0;
  int* table = (int*)(ws + off);       off += (size_t)GRID_SZ * 4;
  int* i133  = (int*)(ws + off);       off += (size_t)9 * NVOX * 4;
  int* i313  = (int*)(ws + off);       off += (size_t)9 * NVOX * 4;
  _Float16* wp1  = (_Float16*)(ws + off); off += (size_t)5 * 2 * 32 * 16 * 2;
  _Float16* wp2  = (_Float16*)(ws + off); off += (size_t)5 * 2 * 32 * 16 * 2;
  _Float16* wp12 = (_Float16*)(ws + off); off += (size_t)9 * 2 * 32 * 16 * 2;
  _Float16* wp3  = (_Float16*)(ws + off); off += (size_t)9 * 2 * 32 * 16 * 2;
  float* bufA = (float*)(ws + off);    off += (size_t)NVOX * 32 * 4;
  float* bufB = (float*)(ws + off);    off += (size_t)NVOX * 32 * 4;

  fill_table_k<<<(GRID_SZ + 255) / 256, 256, 0, stream>>>(table);
  scatter_table_k<<<(NVOX + 255) / 256, 256, 0, stream>>>(coords, table);
  nbidx_k<<<(NVOX + 255) / 256, 256, 0, stream>>>(coords, table, i133, i313);
  pack_w16_k<<<2, 256, 0, stream>>>(w1, wp1);
  pack_w16_k<<<2, 256, 0, stream>>>(w2, wp2);
  pack_w32_k<<<3, 256, 0, stream>>>(w12, wp12);
  pack_w32_k<<<3, 256, 0, stream>>>(w3, wp3);

  int cblocks = (NTILES + 7) / 8;  // 8 waves (tiles) per 256-thread block
  // shortcut path: conv(1,3,3) -> bn0 ; conv(3,1,3) -> bn1
  conv16_k<<<cblocks, 256, 0, stream>>>(feats, i133, wp1, gm, bt, mu, vr, 0, bufA);
  conv32_k<<<cblocks, 256, 0, stream>>>(bufA, i313, wp12, gm, bt, mu, vr, 1, nullptr, bufB);
  // residual path: conv(3,1,3) -> bn2 ; conv(1,3,3) -> bn3 ; fused +shortcut
  conv16_k<<<cblocks, 256, 0, stream>>>(feats, i313, wp2, gm, bt, mu, vr, 2, bufA);
  conv32_k<<<cblocks, 256, 0, stream>>>(bufA, i133, wp3, gm, bt, mu, vr, 3, bufB, (float*)d_out);
}
